// LinAtten_72567767433690
// MI455X (gfx1250) — matrix operand order repaired and checked by reading, not compiled, not run on hardware
//
#include <hip/hip_runtime.h>
#include <hip/hip_fp16.h>

typedef __attribute__((ext_vector_type(16))) _Float16 v16h;
typedef __attribute__((ext_vector_type(8)))  float    v8f;
typedef __attribute__((ext_vector_type(4)))  unsigned u32x4;
typedef __attribute__((ext_vector_type(8)))  unsigned u32x8;

#define DEV __device__ __forceinline__

static constexpr int B_  = 4;
static constexpr int N_  = 4096;
static constexpr int C_  = 1024;
static constexpr int H_  = 16;
static constexpr int D_  = 64;
static constexpr int M_  = B_ * N_;        // 16384 rows for the big GEMMs

DEV v8f wmma32(v16h a, v16h b, v8f c) {
  // D = A(16x32 f16) * B(32x16 f16) + C(16x16 f32)
  return __builtin_amdgcn_wmma_f32_16x16x32_f16(
      /*neg_a=*/false, a, /*neg_b=*/false, b,
      /*c_mod=*/(short)0, c, /*reuse_a=*/false, /*reuse_b=*/false);
}

// ---- fragment loaders (ISA 7.12.2 layouts, wave32) --------------------------
// A 16x32 f16 from row-major tile (stride ld, multiple of 8 halfs):
// lane m = lane&15, hi = lane>>4; VGPR pair p holds k = (p>>2)*16 + hi*8 + (p&3)*2.
// Each lane reads two 16-byte contiguous runs -> 2x ds_load_b128.
DEV v16h frag_a(const _Float16* p, int ld, int lane) {
  const int m = lane & 15, hi = lane >> 4;
  v16h a;
#pragma unroll
  for (int q = 0; q < 8; ++q) {
    const int k = ((q >> 2) << 4) + (hi << 3) + ((q & 3) << 1);
    a[2 * q]     = p[m * ld + k];
    a[2 * q + 1] = p[m * ld + k + 1];
  }
  return a;
}

// B 32x16 f16 from an n-major (transposed) tile Bt[n][k] (stride ld, mult of 8):
// lane n = lane&15 owns K = hi*8 .. hi*8+7 and 16+hi*8 .. 16+hi*8+7 -> 2x ds_load_b128.
DEV v16h frag_bt(const _Float16* p, int ld, int lane) {
  const int n = lane & 15, kb = (lane >> 4) << 3;
  v16h b;
#pragma unroll
  for (int q = 0; q < 16; ++q) b[q] = p[n * ld + kb + ((q >> 3) << 4) + (q & 7)];
  return b;
}

DEV float elu1(float x) { return x >= 0.f ? x + 1.f : __expf(x); }

// Async global->LDS copy of 16 bytes per lane (no VGPR round-trip); ASYNCcnt.
DEV void async_ld_b128(unsigned lds_off, unsigned long long gaddr) {
  asm volatile("global_load_async_to_lds_b128 %0, %1, off"
               :: "v"(lds_off), "v"(gaddr) : "memory");
}
DEV void wait_asynccnt0() { asm volatile("s_wait_asynccnt 0x0" ::: "memory"); }

// ---------------------------------------------------------------------------
// Kernel 1: qkv = x @ W_qkv, fused elu+1 on q,k, scatter to [B,H,N,D] f16.
// grid (M/128, 48), block 256. Block tile 128x64; wave tile 32x32.
// ---------------------------------------------------------------------------
__global__ __launch_bounds__(256)
void la_qkv_gemm(const float* __restrict__ x, const float* __restrict__ Wqkv,
                 _Float16* __restrict__ qf, _Float16* __restrict__ kf,
                 _Float16* __restrict__ vf) {
  __shared__ _Float16 As[128][32 + 8];     // row-major (m,k)
  __shared__ _Float16 Bt[64][32 + 8];      // transposed (n,k) for frag_bt
  const int tid = threadIdx.x, lane = tid & 31, wid = tid >> 5;
  const int m_blk = blockIdx.x * 128;
  const int n_blk = blockIdx.y;              // 0..47
  const int which = n_blk >> 4;              // 0=q 1=k 2=v
  const int h     = n_blk & 15;
  const int wm = (wid & 3) * 32, wn = (wid >> 2) * 32;

  v8f acc[2][2] = {};
  for (int k0 = 0; k0 < C_; k0 += 32) {
    if (k0 + 32 < C_)
      __builtin_prefetch(&x[(size_t)(m_blk + (tid >> 1)) * C_ + k0 + 32], 0, 1);
#pragma unroll 4
    for (int i = tid; i < 128 * 32; i += 256) {
      const int r = i >> 5, c = i & 31;
      As[r][c] = (_Float16)x[(size_t)(m_blk + r) * C_ + k0 + c];
    }
#pragma unroll 2
    for (int i = tid; i < 32 * 64; i += 256) {
      const int r = i >> 6, c = i & 63;    // r = k, c = n (coalesced global read)
      Bt[c][r] = (_Float16)Wqkv[(size_t)(k0 + r) * (3 * C_) + n_blk * 64 + c];
    }
    __syncthreads();
    const v16h a0 = frag_a(&As[wm][0],       40, lane);
    const v16h a1 = frag_a(&As[wm + 16][0],  40, lane);
    const v16h b0 = frag_bt(&Bt[wn][0],      40, lane);
    const v16h b1 = frag_bt(&Bt[wn + 16][0], 40, lane);
    acc[0][0] = wmma32(a0, b0, acc[0][0]);
    acc[0][1] = wmma32(a0, b1, acc[0][1]);
    acc[1][0] = wmma32(a1, b0, acc[1][0]);
    acc[1][1] = wmma32(a1, b1, acc[1][1]);
    __syncthreads();
  }
  _Float16* dst = (which == 0) ? qf : (which == 1) ? kf : vf;
  const int hi = lane >> 4, n16 = lane & 15;
  if (which < 2) {                           // uniform branch: q,k get elu+1
#pragma unroll
    for (int im = 0; im < 2; ++im)
#pragma unroll
      for (int in = 0; in < 2; ++in)
#pragma unroll
        for (int r = 0; r < 8; ++r) {
          const int m = m_blk + wm + im * 16 + hi * 8 + r;
          const int d = wn + in * 16 + n16;
          const float v = elu1(acc[im][in][r]);
          const int b = m >> 12, ntok = m & (N_ - 1);
          dst[(((size_t)(b * H_ + h)) * N_ + ntok) * D_ + d] = (_Float16)v;
        }
  } else {
#pragma unroll
    for (int im = 0; im < 2; ++im)
#pragma unroll
      for (int in = 0; in < 2; ++in)
#pragma unroll
        for (int r = 0; r < 8; ++r) {
          const int m = m_blk + wm + im * 16 + hi * 8 + r;
          const int d = wn + in * 16 + n16;
          const int b = m >> 12, ntok = m & (N_ - 1);
          dst[(((size_t)(b * H_ + h)) * N_ + ntok) * D_ + d] =
              (_Float16)acc[im][in][r];
        }
  }
}

// ---------------------------------------------------------------------------
// Kernel 2: kv[bh] += k^T v over a 512-token slab; ksum[bh] += sum_n k.
// grid (B*H, N/512), block 256. Wave tile 16x32 (2 WMMA). f32 atomics into ws.
// Both K and V tiles staged transposed so all fragment loads are b128.
// ---------------------------------------------------------------------------
__global__ __launch_bounds__(256)
void la_kv_state(const _Float16* __restrict__ kf, const _Float16* __restrict__ vf,
                 float* __restrict__ kv32, float* __restrict__ ksum) {
  __shared__ _Float16 Kt[64][32 + 8];      // (d, n): A = k^T row-major
  __shared__ _Float16 Vt[64][32 + 8];      // (e, n): B tile n-major for frag_bt
  __shared__ float red[4][64];
  const int tid = threadIdx.x, lane = tid & 31, wid = tid >> 5;
  const int bh = blockIdx.x;
  const int n_base = blockIdx.y * 512;
  const _Float16* kp = kf + (size_t)bh * N_ * D_;
  const _Float16* vp = vf + (size_t)bh * N_ * D_;
  const int wm = (wid >> 1) * 16;            // d-tile
  const int wn = (wid & 1) * 32;             // e-offset
  const int d_red = tid & 63, q4 = tid >> 6;

  v8f acc[2] = {};
  float ks = 0.f;
  for (int n0 = n_base; n0 < n_base + 512; n0 += 32) {
#pragma unroll 2
    for (int i = tid; i < 32 * 64; i += 256) {
      const int r = i >> 6, c = i & 63;    // r = n, c = d/e (coalesced global)
      Kt[c][r] = kp[(size_t)(n0 + r) * D_ + c];
      Vt[c][r] = vp[(size_t)(n0 + r) * D_ + c];
    }
    __syncthreads();
    const v16h a  = frag_a(&Kt[wm][0],       40, lane);  // A[d][n]
    const v16h b0 = frag_bt(&Vt[wn][0],      40, lane);
    const v16h b1 = frag_bt(&Vt[wn + 16][0], 40, lane);
    acc[0] = wmma32(a, b0, acc[0]);
    acc[1] = wmma32(a, b1, acc[1]);
#pragma unroll
    for (int r = 0; r < 8; ++r) ks += (float)Kt[d_red][q4 * 8 + r];
    __syncthreads();
  }
  red[q4][d_red] = ks;
  __syncthreads();
  if (tid < 64)
    atomicAdd(&ksum[bh * D_ + tid],
              red[0][tid] + red[1][tid] + red[2][tid] + red[3][tid]);
  const int hi = lane >> 4, n16 = lane & 15;
#pragma unroll
  for (int in = 0; in < 2; ++in)
#pragma unroll
    for (int r = 0; r < 8; ++r) {
      const int d = wm + hi * 8 + r;
      const int e = wn + in * 16 + n16;
      atomicAdd(&kv32[(size_t)bh * D_ * D_ + d * D_ + e], acc[in][r]);
    }
}

// ---------------------------------------------------------------------------
// Kernel 3: out = (q @ kv) * z, scatter to attn[B,N,C] f16.
// grid (B*H, N/128), block 256. Wave tile 16x64; K=64 (2 WMMA K-steps).
// Q tile is fetched by the Tensor Data Mover: one descriptor copies the
// 64x128 f16 tile (stride 64) into LDS, inserting 16B of pad per 128B row to
// reproduce the [128][72] padded layout. Tracked by TENSORcnt.
// ---------------------------------------------------------------------------
__global__ __launch_bounds__(256)
void la_apply(const _Float16* __restrict__ qf, const float* __restrict__ kv32,
              const float* __restrict__ ksum, _Float16* __restrict__ attn) {
  __shared__ _Float16 Qs[128][64 + 8];     // row-major (m,k), 144B row stride
  __shared__ _Float16 KVt[64][64 + 8];     // (e, d): B tile n-major
  __shared__ float zs[128];
  __shared__ float kss[64];
  const int tid = threadIdx.x, lane = tid & 31, wid = tid >> 5;
  const int bh = blockIdx.x, b = bh >> 4, h = bh & 15;
  const int n0 = blockIdx.y * 128;
  const _Float16* qp = qf + (size_t)bh * N_ * D_;

  if (wid == 0) {
    // TDM descriptor: 2D tensor, data_size=2B, tile 64(x) x 128(y), stride 64,
    // LDS padding 4 DWORDs after every 32 DWORDs (-> row stride 72 halfs).
    const unsigned long long ga =
        (unsigned long long)(uintptr_t)(qp + (size_t)n0 * D_);
    u32x4 g0;
    g0[0] = 1u;                                       // count=1 valid descriptor
    g0[1] = (unsigned)(uintptr_t)&Qs[0][0];           // lds_addr
    g0[2] = (unsigned)ga;                             // global_addr[31:0]
    g0[3] = (unsigned)(ga >> 32) | (2u << 30);        // addr[56:32] | type=2
    u32x8 g1;
    g1[0] = (1u << 16)          // data_size: 2 bytes
          | (1u << 20)          // pad_enable
          | (4u << 22)          // pad_interval: 32 DWORDs (=128B of data)
          | (3u << 25);         // pad_amount: 4 DWORDs (=16B = 8 halfs)
    g1[1] = (unsigned)(D_) << 16;        // tensor_dim0 = 64 (lo16)
    g1[2] = (unsigned)(N_) << 16;        // dim0 hi=0 | tensor_dim1 lo16 = 4096
    g1[3] = (unsigned)(D_) << 16;        // dim1 hi=0 | tile_dim0 = 64
    g1[4] = 128u;                        // tile_dim1 = 128, tile_dim2 = 0
    g1[5] = (unsigned)(D_);              // tensor_dim0_stride = 64 (lo32)
    g1[6] = 0u;                          // stride0 hi | stride1 lo
    g1[7] = 0u;
    asm volatile("tensor_load_to_lds %0, %1" :: "s"(g0), "s"(g1) : "memory");
  }
#pragma unroll 2
  for (int i = tid; i < 64 * 64; i += 256) {
    const int d = i >> 6, e = i & 63;      // coalesced global read over e
    KVt[e][d] = (_Float16)kv32[(size_t)bh * D_ * D_ + i];
  }
  if (tid < 64) kss[tid] = ksum[bh * D_ + tid];
  if (wid == 0) __builtin_amdgcn_s_wait_tensorcnt(0);
  __syncthreads();

  {  // z = 1/(q . ksum + eps); lane pairs split the 64-dim dot
    const int t = tid >> 1, half = (tid & 1) * 32;
    float s = 0.f;
#pragma unroll
    for (int d = 0; d < 32; ++d) s += (float)Qs[t][half + d] * kss[half + d];
    s += __shfl_xor(s, 1, 32);
    if ((tid & 1) == 0) zs[t] = 1.f / (s + 1e-6f);
  }
  __syncthreads();

  const int wm = wid * 16;
  v8f acc[4] = {};
#pragma unroll
  for (int k0 = 0; k0 < 64; k0 += 32) {
    const v16h a = frag_a(&Qs[wm][k0], 72, lane);
#pragma unroll
    for (int j = 0; j < 4; ++j) {
      const v16h bfr = frag_bt(&KVt[j * 16][k0], 72, lane);
      acc[j] = wmma32(a, bfr, acc[j]);
    }
  }
  const int hi = lane >> 4, n16 = lane & 15;
#pragma unroll
  for (int j = 0; j < 4; ++j)
#pragma unroll
    for (int r = 0; r < 8; ++r) {
      const int m = wm + hi * 8 + r;         // token within block
      const int e = j * 16 + n16;
      const float v = acc[j][r] * zs[m];
      attn[((size_t)(b * N_ + n0 + m)) * C_ + h * D_ + e] = (_Float16)v;
    }
}

// ---------------------------------------------------------------------------
// Kernel 4: out = attn @ W_proj + b_proj (f32 out).
// grid (M/128, 16), block 256. A tile staged with async global->LDS b128
// copies (ASYNCcnt), bypassing VGPRs; W tile converts f32->f16 via VGPRs.
// ---------------------------------------------------------------------------
__global__ __launch_bounds__(256)
void la_proj(const _Float16* __restrict__ attn, const float* __restrict__ Wp,
             const float* __restrict__ bias, float* __restrict__ out) {
  __shared__ _Float16 As[128][32 + 8];
  __shared__ _Float16 Bt[64][32 + 8];
  const int tid = threadIdx.x, lane = tid & 31, wid = tid >> 5;
  const int m_blk = blockIdx.x * 128;
  const int n_blk = blockIdx.y;              // 0..15
  const int wm = (wid & 3) * 32, wn = (wid >> 2) * 32;

  v8f acc[2][2] = {};
  for (int k0 = 0; k0 < C_; k0 += 32) {
    // A tile: 128 rows x 32 halfs = 512 16B chunks, async DMA to LDS
#pragma unroll
    for (int ch = tid; ch < 128 * 4; ch += 256) {
      const int r = ch >> 2, cc = (ch & 3) * 8;
      async_ld_b128(
          (unsigned)(uintptr_t)&As[r][cc],
          (unsigned long long)(uintptr_t)&attn[(size_t)(m_blk + r) * C_ + k0 + cc]);
    }
#pragma unroll 2
    for (int i = tid; i < 32 * 64; i += 256) {
      const int r = i >> 6, c = i & 63;
      Bt[c][r] = (_Float16)Wp[(size_t)(k0 + r) * C_ + n_blk * 64 + c];
    }
    wait_asynccnt0();
    __syncthreads();
    const v16h a0 = frag_a(&As[wm][0],       40, lane);
    const v16h a1 = frag_a(&As[wm + 16][0],  40, lane);
    const v16h b0 = frag_bt(&Bt[wn][0],      40, lane);
    const v16h b1 = frag_bt(&Bt[wn + 16][0], 40, lane);
    acc[0][0] = wmma32(a0, b0, acc[0][0]);
    acc[0][1] = wmma32(a0, b1, acc[0][1]);
    acc[1][0] = wmma32(a1, b0, acc[1][0]);
    acc[1][1] = wmma32(a1, b1, acc[1][1]);
    __syncthreads();
  }
  const int hi = lane >> 4, n16 = lane & 15;
#pragma unroll
  for (int im = 0; im < 2; ++im)
#pragma unroll
    for (int in = 0; in < 2; ++in) {
      const int n = n_blk * 64 + wn + in * 16 + n16;
      const float bv = bias[n];
#pragma unroll
      for (int r = 0; r < 8; ++r) {
        const int m = m_blk + wm + im * 16 + hi * 8 + r;
        out[(size_t)m * C_ + n] = acc[im][in][r] + bv;
      }
    }
}

// ---------------------------------------------------------------------------
extern "C" void kernel_launch(void* const* d_in, const int* in_sizes, int n_in,
                              void* d_out, int out_size, void* d_ws, size_t ws_size,
                              hipStream_t stream) {
  const float* x     = (const float*)d_in[0];
  const float* Wqkv  = (const float*)d_in[1];
  const float* Wproj = (const float*)d_in[2];
  const float* bproj = (const float*)d_in[3];
  float* out = (float*)d_out;

  char* ws = (char*)d_ws;
  const size_t qkv_bytes  = (size_t)B_ * H_ * N_ * D_ * sizeof(_Float16); // 32 MB
  const size_t attn_bytes = (size_t)M_ * C_ * sizeof(_Float16);           // 32 MB
  _Float16* qf   = (_Float16*)(ws);
  _Float16* kf   = (_Float16*)(ws + qkv_bytes);
  _Float16* vf   = (_Float16*)(ws + 2 * qkv_bytes);
  _Float16* attn = (_Float16*)(ws + 3 * qkv_bytes);
  float*    kv32 = (float*)(ws + 3 * qkv_bytes + attn_bytes);
  float*    ksum = kv32 + (size_t)B_ * H_ * D_ * D_;
  const size_t zero_bytes =
      ((size_t)B_ * H_ * D_ * D_ + (size_t)B_ * H_ * D_) * sizeof(float);

  // accumulators must be zeroed every call (graph-capturable memset node)
  hipMemsetAsync(kv32, 0, zero_bytes, stream);

  la_qkv_gemm<<<dim3(M_ / 128, 48), 256, 0, stream>>>(x, Wqkv, qf, kf, vf);
  la_kv_state<<<dim3(B_ * H_, N_ / 512), 256, 0, stream>>>(kf, vf, kv32, ksum);
  la_apply<<<dim3(B_ * H_, N_ / 128), 256, 0, stream>>>(qf, kv32, ksum, attn);
  la_proj<<<dim3(M_ / 128, 16), 256, 0, stream>>>(attn, Wproj, bproj, out);
}
